// Transformer_Encoder_24627342475429
// MI455X (gfx1250) — matrix operand order repaired and checked by reading, not compiled, not run on hardware
//
#include <hip/hip_runtime.h>
#include <cmath>

// ---------------------------------------------------------------------------
// CDNA5 (gfx1250) Swin encoder. wave32, WMMA f32_16x16x32_f16 for all GEMMs
// and attention; weight tiles staged to LDS with GLOBAL_LOAD_ASYNC_TO_LDS_B128
// (ASYNCcnt-tracked, double-buffered). Fragment layouts used here (C/D as in CDNA5 ISA 7.12.2):
//   A (16x32 f16): lane L (m = L&15, half = L>>4):
//     a[0..15] = A[m][half*16 + 0..15] (contiguous k)
//   B (32x16 f16): lane L (n = L&15, kb = (L>>4)*16): b[i] = B[kb+i][n] (contiguous k, not ISA table)
//   C/D (16x16 f32): lane L col n = L&15; vgpr v row m = v + 8*(L>>4)
// ---------------------------------------------------------------------------

typedef __attribute__((ext_vector_type(16))) _Float16 v16h;
typedef __attribute__((ext_vector_type(8)))  _Float16 v8h;
typedef __attribute__((ext_vector_type(8)))  float    v8f;

static __device__ __forceinline__ v16h frag_a(const _Float16* p) {
  // a[0..15] = p[0..15]
  v8h lo = *(const v8h*)p;
  v8h hi = *(const v8h*)(p + 8);
  v16h a;
#pragma unroll
  for (int i = 0; i < 8; ++i) { a[i] = lo[i]; a[i + 8] = hi[i]; }
  return a;
}
static __device__ __forceinline__ v16h frag_b16(const _Float16* p) {
  // 16 contiguous halves
  v8h lo = *(const v8h*)p;
  v8h hi = *(const v8h*)(p + 8);
  v16h b;
#pragma unroll
  for (int i = 0; i < 8; ++i) { b[i] = lo[i]; b[i + 8] = hi[i]; }
  return b;
}

// ---------------------------------------------------------------------------
// Utility kernels
// ---------------------------------------------------------------------------
__global__ void k_f32_to_f16(const float* __restrict__ s, _Float16* __restrict__ d, long n) {
  long i = (long)blockIdx.x * blockDim.x + threadIdx.x;
  if (i < n) d[i] = (_Float16)s[i];
}

__global__ void k_pad_w(const float* __restrict__ s, _Float16* __restrict__ d,
                        int N, int Ks, int Kd) {
  long i = (long)blockIdx.x * blockDim.x + threadIdx.x;
  if (i >= (long)N * Kd) return;
  int k = (int)(i % Kd); int n = (int)(i / Kd);
  d[i] = (k < Ks) ? (_Float16)s[(long)n * Ks + k] : (_Float16)0.f;
}

__global__ void k_add(float* __restrict__ x, const float* __restrict__ o, long n) {
  long i = (long)blockIdx.x * blockDim.x + threadIdx.x;
  if (i < n) x[i] += o[i];
}

// im2col for patch embed: x (8,69,128,256) -> (8*64*128, 288) f16, K padded 276->288
__global__ void k_im2col(const float* __restrict__ x, _Float16* __restrict__ out) {
  long idx = (long)blockIdx.x * blockDim.x + threadIdx.x;
  const long total = (long)8 * 64 * 128 * 288;
  if (idx >= total) return;
  int k = (int)(idx % 288); long r = idx / 288;
  int wp = (int)(r % 128); int hp = (int)((r / 128) % 64); int b = (int)(r / (128 * 64));
  _Float16 v = (_Float16)0.f;
  if (k < 276) {
    int c = k >> 2, ph = (k >> 1) & 1, pw = k & 1;
    v = (_Float16)x[(((long)b * 69 + c) * 128 + hp * 2 + ph) * 256 + wp * 2 + pw];
  }
  out[idx] = v;
}

// patch merging concat: X (8,H,W,C) -> O (8,H/2,W/2,4C), order [x00,x10,x01,x11]
__global__ void k_merge(const float* __restrict__ X, float* __restrict__ O,
                        int H, int Wd, int C) {
  long idx = (long)blockIdx.x * blockDim.x + threadIdx.x;
  long total = (long)8 * (H / 2) * (Wd / 2) * 4 * C;
  if (idx >= total) return;
  int c4 = (int)(idx % (4 * C)); long r = idx / (4 * C);
  int w2 = (int)(r % (Wd / 2)); int h2 = (int)((r / (Wd / 2)) % (H / 2));
  int b  = (int)(r / ((long)(Wd / 2) * (H / 2)));
  int j = c4 / C, c = c4 % C;
  int dh = j & 1, dw = j >> 1;  // j: 0->(0,0) 1->(1,0) 2->(0,1) 3->(1,1)
  O[idx] = X[(((long)b * H + h2 * 2 + dh) * Wd + w2 * 2 + dw) * C + c];
}

// LayerNorm; optional fused roll(-shift)+window-partition gather of the source
// token; output f16 (outH) or f32 (outF), row-major (T, C).
__global__ void k_ln(const float* __restrict__ X, const float* __restrict__ w,
                     const float* __restrict__ b, _Float16* __restrict__ outH,
                     float* __restrict__ outF, int T, int C,
                     int winOrder, int H, int Wd, int shift) {
  int r = blockIdx.x * blockDim.x + threadIdx.x;
  if (r >= T) return;
  const float* src;
  if (winOrder) {
    int tokPerImg = H * Wd;
    int bi = r / tokPerImg, rem = r % tokPerImg;
    int Wg = Wd >> 3;
    int wid = rem >> 6, t = rem & 63;
    int wr = wid / Wg, wc = wid % Wg;
    int hh = wr * 8 + (t >> 3), ww = wc * 8 + (t & 7);
    int h0 = (hh + shift) % H, w0 = (ww + shift) % Wd;
    src = X + ((long)(bi * H + h0) * Wd + w0) * C;
  } else {
    src = X + (long)r * C;
  }
  float mu = 0.f;
  for (int c = 0; c < C; ++c) mu += src[c];
  mu /= C;
  float var = 0.f;
  for (int c = 0; c < C; ++c) { float d = src[c] - mu; var += d * d; }
  var /= C;
  float inv = rsqrtf(var + 1e-5f);
  if (outH) {
    _Float16* dst = outH + (long)r * C;
    for (int c = 0; c < C; ++c) dst[c] = (_Float16)((src[c] - mu) * inv * w[c] + b[c]);
  } else {
    float* dst = outF + (long)r * C;
    for (int c = 0; c < C; ++c) dst[c] = (src[c] - mu) * inv * w[c] + b[c];
  }
}

// window-reverse + roll(+shift) + residual add: X[b,h0,w0,:] += O[winrow,:]
__global__ void k_winrev_add(const float* __restrict__ O, float* __restrict__ X,
                             int T, int C, int H, int Wd, int shift) {
  long idx = (long)blockIdx.x * blockDim.x + threadIdx.x;
  if (idx >= (long)T * C) return;
  int c = (int)(idx % C); long r = idx / C;
  int tokPerImg = H * Wd;
  int bi = (int)(r / tokPerImg); int rem = (int)(r % tokPerImg);
  int Wg = Wd >> 3;
  int wid = rem >> 6, t = rem & 63;
  int wr = wid / Wg, wc = wid % Wg;
  int hh = wr * 8 + (t >> 3), ww = wc * 8 + (t & 7);
  int h0 = (hh + shift) % H, w0 = (ww + shift) % Wd;
  X[((long)(bi * H + h0) * Wd + w0) * C + c] += O[idx];
}

// ---------------------------------------------------------------------------
// WMMA GEMM: D[M,N] = act(A[M,K] x W[N,K]^T + bias), optional q-scale on the
// first scaleCols columns, optional exact GELU, f16 or f32 store.
// block = 256 thr (8 waves) computing 128x64. The 64x32-half weight tile for
// each k-step is staged to LDS once per block with GLOBAL_LOAD_ASYNC_TO_LDS_
// B128 (each thread copies 16B; one async op per wave), double-buffered so the
// DMA of step k+1 overlaps the WMMAs of step k. ASYNCcnt + barrier sync.
// Requires M % 128 == 0 (true for every GEMM in this model).
// ---------------------------------------------------------------------------
__global__ void __launch_bounds__(256)
k_gemm(const _Float16* __restrict__ A, const _Float16* __restrict__ Wt,
       const float* __restrict__ bias, float* __restrict__ Df,
       _Float16* __restrict__ Dh, int M, int N, int K,
       int gelu, float scale, int scaleCols) {
  __shared__ __align__(16) _Float16 Wl[2][64 * 32];  // 2 x 4KB weight tiles
  int tid = threadIdx.x;
  int wave = tid >> 5, lane = tid & 31;
  int lm = lane & 15, lh = lane >> 4;
  int m0 = (blockIdx.y * 8 + wave) * 16;
  int n0 = blockIdx.x * 64;

  // per-thread 16B slice of the 64x32-half weight tile
  int srow = tid >> 2, sseg = tid & 3;
  const _Float16* gsrc0 = Wt + (size_t)(n0 + srow) * K + sseg * 8;
  // LDS byte address = low 32 bits of the generic shared pointer (ISA 10.2)
  unsigned ldsbase = (unsigned)(size_t)(&Wl[0][0]) + (unsigned)(srow * 64 + sseg * 16);

  auto stage = [&](int buf, int k0) {
    unsigned lds = ldsbase + (unsigned)(buf * 4096);
    unsigned long long g = (unsigned long long)(size_t)(gsrc0 + k0);
    asm volatile("global_load_async_to_lds_b128 %0, %1, off"
                 :: "v"(lds), "v"(g) : "memory");
  };

  v8f c[4];
#pragma unroll
  for (int t = 0; t < 4; ++t)
#pragma unroll
    for (int i = 0; i < 8; ++i) c[t][i] = 0.f;

  const _Float16* arow = A + (size_t)(m0 + lm) * K;

  stage(0, 0);
  int buf = 0;
  for (int k0 = 0; k0 < K; k0 += 32) {
    asm volatile("s_wait_asynccnt 0x0" ::: "memory");  // own DMA done
    __syncthreads();                                   // everyone's DMA done
    if (k0 + 32 < K) stage(buf ^ 1, k0 + 32);          // prefetch next tile
    v16h a = frag_a(arow + k0 + lh * 16);
    const _Float16* wb = &Wl[buf][0];
#pragma unroll
    for (int t = 0; t < 4; ++t) {
      v16h b = frag_b16(wb + (t * 16 + lm) * 32 + lh * 16);
      c[t] = __builtin_amdgcn_wmma_f32_16x16x32_f16(false, a, false, b,
                                                    (short)0, c[t], false, false);
    }
    __syncthreads();   // all reads of tile[buf] done before it is restaged
    buf ^= 1;
  }

#pragma unroll
  for (int t = 0; t < 4; ++t) {
    int col = n0 + t * 16 + lm;
    float bv = bias ? bias[col] : 0.f;
#pragma unroll
    for (int v = 0; v < 8; ++v) {
      int row = m0 + v + 8 * lh;
      float xv = c[t][v] + bv;
      if (col < scaleCols) xv *= scale;
      if (gelu) xv = 0.5f * xv * (1.f + erff(xv * 0.70710678118654752f));
      if (Dh) Dh[(long)row * N + col] = (_Float16)xv;
      else    Df[(long)row * N + col] = xv;
    }
  }
}

// ---------------------------------------------------------------------------
// Windowed attention, one wave per (window, head). head_dim == 32 always.
// qkv (winrow, 3C) f16 with q pre-scaled; out (winrow, C) f16, heads interleaved.
// Relative-position bias and shifted-window mask computed arithmetically.
// ---------------------------------------------------------------------------
__global__ void __launch_bounds__(64)
k_attn(const _Float16* __restrict__ qkv, const float* __restrict__ rpb,
       _Float16* __restrict__ outA, int nWin, int nh, int C,
       int Hg, int Wg, int shift) {
  __shared__ __align__(16) float    Ssh[2][64 * 64];
  __shared__ __align__(16) _Float16 Psh[2][64 * 64];
  int wave = threadIdx.x >> 5, lane = threadIdx.x & 31;
  int task = blockIdx.x * 2 + wave;
  if (task >= nWin * nh) return;
  int w = task / nh, h = task % nh;
  int lm = lane & 15, lh = lane >> 4;
  int C3 = 3 * C;
  const _Float16* Q  = qkv + (long)w * 64 * C3 + h * 32;
  const _Float16* Kp = Q + C;
  const _Float16* Vp = Q + 2 * C;
  float*    S = Ssh[wave];
  _Float16* P = Psh[wave];

  int wid = w % (Hg * Wg);
  int wr = wid / Wg, wc = wid % Wg;
  int Ht = Hg * 8, Wt = Wg * 8;

  // S = Q K^T + bias (+mask)
#pragma unroll
  for (int mi = 0; mi < 4; ++mi) {
    v16h a = frag_a(Q + (long)(mi * 16 + lm) * C3 + lh * 16);
#pragma unroll
    for (int ni = 0; ni < 4; ++ni) {
      v16h b = frag_b16(Kp + (long)(ni * 16 + lm) * C3 + lh * 16);
      v8f c;
#pragma unroll
      for (int i = 0; i < 8; ++i) c[i] = 0.f;
      c = __builtin_amdgcn_wmma_f32_16x16x32_f16(false, a, false, b,
                                                 (short)0, c, false, false);
      int col = ni * 16 + lm;
      int kr = col >> 3, kc = col & 7;
#pragma unroll
      for (int v = 0; v < 8; ++v) {
        int row = mi * 16 + v + 8 * lh;
        int qr = row >> 3, qc = row & 7;
        int ridx = (qr - kr + 7) * 15 + (qc - kc + 7);
        float val = c[v] + rpb[ridx * nh + h];
        if (shift) {
          int gh = wr * 8 + qr, gw = wc * 8 + qc;
          int lq = (gh < Ht - 8 ? 0 : (gh < Ht - 4 ? 1 : 2)) * 3 +
                   (gw < Wt - 8 ? 0 : (gw < Wt - 4 ? 1 : 2));
          int gh2 = wr * 8 + kr, gw2 = wc * 8 + kc;
          int lk = (gh2 < Ht - 8 ? 0 : (gh2 < Ht - 4 ? 1 : 2)) * 3 +
                   (gw2 < Wt - 8 ? 0 : (gw2 < Wt - 4 ? 1 : 2));
          if (lq != lk) val -= 100.f;
        }
        S[row * 64 + col] = val;
      }
    }
  }

  // row softmax (2 rows per lane), write P as f16
#pragma unroll
  for (int rr = 0; rr < 2; ++rr) {
    int row = lane + rr * 32;
    float mx = -1e30f;
    for (int j = 0; j < 64; ++j) mx = fmaxf(mx, S[row * 64 + j]);
    float sum = 0.f;
    for (int j = 0; j < 64; ++j) {
      float e = expf(S[row * 64 + j] - mx);
      S[row * 64 + j] = e; sum += e;
    }
    float inv = 1.f / sum;
    for (int j = 0; j < 64; ++j) P[row * 64 + j] = (_Float16)(S[row * 64 + j] * inv);
  }

  // O = P V  (K = 64 in two 32-steps)
#pragma unroll
  for (int mi = 0; mi < 4; ++mi) {
#pragma unroll
    for (int ni = 0; ni < 2; ++ni) {
      v8f c;
#pragma unroll
      for (int i = 0; i < 8; ++i) c[i] = 0.f;
#pragma unroll
      for (int ks = 0; ks < 2; ++ks) {
        v16h a = frag_a(P + (mi * 16 + lm) * 64 + ks * 32 + lh * 16);
        v16h b;
#pragma unroll
        for (int i = 0; i < 16; ++i)
          b[i] = Vp[(long)(ks * 32 + lh * 16 + i) * C3 + ni * 16 + lm];
        c = __builtin_amdgcn_wmma_f32_16x16x32_f16(false, a, false, b,
                                                   (short)0, c, false, false);
      }
#pragma unroll
      for (int v = 0; v < 8; ++v) {
        int row = mi * 16 + v + 8 * lh;
        outA[(long)(w * 64 + row) * C + h * 32 + ni * 16 + lm] = (_Float16)c[v];
      }
    }
  }
}

// ---------------------------------------------------------------------------
// Host orchestration
// ---------------------------------------------------------------------------
extern "C" void kernel_launch(void* const* d_in, const int* in_sizes, int n_in,
                              void* d_out, int out_size, void* d_ws, size_t ws_size,
                              hipStream_t stream) {
  (void)in_sizes; (void)n_in; (void)out_size; (void)ws_size;

  // ---- parameter pointers -------------------------------------------------
  // setup_inputs() order: x first, then params flattened jax-pytree style
  // (dict keys sorted alphabetically, lists in order).
  const float* x = (const float*)d_in[0];
  int gi = 1;
  const float* fn_b = (const float*)d_in[gi++];
  const float* fn_w = (const float*)d_in[gi++];
  struct Blk {
    const float *fc1_b, *fc1_w, *fc2_b, *fc2_w, *n1_b, *n1_w, *n2_b, *n2_w,
                *proj_b, *proj_w, *qkv_b, *qkv_w, *rpb;
  };
  Blk blk[3][2];
  const float *dn_nb[3] = {0, 0, 0}, *dn_nw[3] = {0, 0, 0}, *dn_rw[3] = {0, 0, 0};
  for (int L = 0; L < 3; ++L) {
    for (int j = 0; j < 2; ++j) {
      Blk& B = blk[L][j];
      B.fc1_b  = (const float*)d_in[gi++]; B.fc1_w  = (const float*)d_in[gi++];
      B.fc2_b  = (const float*)d_in[gi++]; B.fc2_w  = (const float*)d_in[gi++];
      B.n1_b   = (const float*)d_in[gi++]; B.n1_w   = (const float*)d_in[gi++];
      B.n2_b   = (const float*)d_in[gi++]; B.n2_w   = (const float*)d_in[gi++];
      B.proj_b = (const float*)d_in[gi++]; B.proj_w = (const float*)d_in[gi++];
      B.qkv_b  = (const float*)d_in[gi++]; B.qkv_w  = (const float*)d_in[gi++];
      B.rpb    = (const float*)d_in[gi++];
    }
    if (L > 0) {
      dn_nb[L] = (const float*)d_in[gi++];
      dn_nw[L] = (const float*)d_in[gi++];
      dn_rw[L] = (const float*)d_in[gi++];
    }
  }
  const float* pe_b  = (const float*)d_in[gi++];
  const float* pe_nb = (const float*)d_in[gi++];
  const float* pe_nw = (const float*)d_in[gi++];
  const float* pe_w  = (const float*)d_in[gi++];

  // ---- workspace arena ----------------------------------------------------
  size_t off = 0;
  auto arena = [&](size_t bytes) -> void* {
    void* p = (char*)d_ws + off;
    off = (off + bytes + 255) & ~(size_t)255;
    return p;
  };
  float*    X  = (float*)   arena((size_t)65536 * 192 * 4);  // residual stream
  float*    Of = (float*)   arena((size_t)65536 * 192 * 4);  // f32 GEMM out
  _Float16* Ah = (_Float16*)arena((size_t)65536 * 288 * 2);  // f16 act A
  _Float16* Bh = (_Float16*)arena((size_t)65536 * 768 * 2);  // f16 act B (qkv/gelu)

  auto conv = [&](const float* s, size_t n) -> _Float16* {
    _Float16* d = (_Float16*)arena(n * 2);
    k_f32_to_f16<<<(unsigned)((n + 255) / 256), 256, 0, stream>>>(s, d, (long)n);
    return d;
  };

  const int Hs[3] = {64, 32, 16}, Wsz[3] = {128, 64, 32};
  const int Cs[3] = {192, 384, 768}, NHs[3] = {6, 12, 24};

  _Float16 *qkvW[3][2], *projW[3][2], *fc1W[3][2], *fc2W[3][2], *redW[3] = {0, 0, 0};
  for (int L = 0; L < 3; ++L) {
    int C = Cs[L];
    for (int j = 0; j < 2; ++j) {
      qkvW[L][j] = conv(blk[L][j].qkv_w, (size_t)3 * C * C);
      projW[L][j] = conv(blk[L][j].proj_w, (size_t)C * C);
      fc1W[L][j] = conv(blk[L][j].fc1_w, (size_t)4 * C * C);
      fc2W[L][j] = conv(blk[L][j].fc2_w, (size_t)C * 4 * C);
    }
    if (L > 0) redW[L] = conv(dn_rw[L], (size_t)C * 2 * C);
  }
  _Float16* peW = (_Float16*)arena((size_t)192 * 288 * 2);
  k_pad_w<<<(192 * 288 + 255) / 256, 256, 0, stream>>>(pe_w, peW, 192, 276, 288);

  auto gemm = [&](const _Float16* A, const _Float16* Wt, const float* bias,
                  float* Df, _Float16* Dh, int M, int N, int K,
                  int gelu, float scale, int scaleCols) {
    dim3 g(N / 64, M / 128);  // M is a multiple of 128 for every GEMM here
    k_gemm<<<g, 256, 0, stream>>>(A, Wt, bias, Df, Dh, M, N, K, gelu, scale, scaleCols);
  };
  auto ln = [&](const float* Xi, const float* w, const float* b, _Float16* oH,
                float* oF, int T, int C, int winOrder, int H, int Wd, int shift) {
    k_ln<<<(T + 255) / 256, 256, 0, stream>>>(Xi, w, b, oH, oF, T, C, winOrder, H, Wd, shift);
  };

  // ---- patch embed: im2col + GEMM + LN -----------------------------------
  {
    long tot = (long)8 * 64 * 128 * 288;
    k_im2col<<<(unsigned)((tot + 255) / 256), 256, 0, stream>>>(x, Ah);
    gemm(Ah, peW, pe_b, Of, nullptr, 65536, 192, 288, 0, 1.f, 0);
    ln(Of, pe_nw, pe_nb, nullptr, X, 65536, 192, 0, 0, 0, 0);
  }

  // ---- output layout (f32): y, d0, d1, d2 --------------------------------
  float* out_f = (float*)d_out;
  const size_t Y_OFF = 0, D_OFF[3] = {3145728, 15728640, 22020096};

  const float qscale = 0.17677669529663687f;  // 1/sqrt(32)

  for (int L = 0; L < 3; ++L) {
    int H = Hs[L], Wd = Wsz[L], C = Cs[L], nh = NHs[L];
    int T = 8 * H * Wd;
    if (L > 0) {
      int pH = Hs[L - 1], pW = Wsz[L - 1], pC = Cs[L - 1];
      long tot = (long)T * 2 * C;  // T * 4*pC
      k_merge<<<(unsigned)((tot + 255) / 256), 256, 0, stream>>>(X, Of, pH, pW, pC);
      ln(Of, dn_nw[L], dn_nb[L], Ah, nullptr, T, 2 * C, 0, 0, 0, 0);
      gemm(Ah, redW[L], nullptr, X, nullptr, T, C, 2 * C, 0, 1.f, 0);
    }
    for (int j = 0; j < 2; ++j) {
      const Blk& B = blk[L][j];
      int shift = (j % 2 == 0) ? 0 : 4;
      int nWin = T / 64, Hg = H / 8, Wg = Wd / 8;
      // attention half
      ln(X, B.n1_w, B.n1_b, Ah, nullptr, T, C, 1, H, Wd, shift);
      gemm(Ah, qkvW[L][j], B.qkv_b, nullptr, Bh, T, 3 * C, C, 0, qscale, C);
      k_attn<<<(nWin * nh + 1) / 2, 64, 0, stream>>>(Bh, B.rpb, Ah, nWin, nh, C,
                                                     Hg, Wg, shift);
      gemm(Ah, projW[L][j], B.proj_b, Of, nullptr, T, C, C, 0, 1.f, 0);
      k_winrev_add<<<(unsigned)(((long)T * C + 255) / 256), 256, 0, stream>>>(
          Of, X, T, C, H, Wd, shift);
      // MLP half
      ln(X, B.n2_w, B.n2_b, Ah, nullptr, T, C, 0, 0, 0, 0);
      gemm(Ah, fc1W[L][j], B.fc1_b, nullptr, Bh, T, 4 * C, C, 1, 1.f, 0);
      gemm(Bh, fc2W[L][j], B.fc2_b, Of, nullptr, T, C, 4 * C, 0, 1.f, 0);
      k_add<<<(unsigned)(((long)T * C + 255) / 256), 256, 0, stream>>>(X, Of, (long)T * C);
    }
    hipMemcpyAsync(out_f + D_OFF[L], X, (size_t)T * C * 4,
                   hipMemcpyDeviceToDevice, stream);
  }

  // ---- final LN -> y ------------------------------------------------------
  ln(X, fn_w, fn_b, nullptr, out_f + Y_OFF, 4096, 768, 0, 0, 0, 0);
}
